// AGBF_4088808866129
// MI455X (gfx1250) — matrix operand order repaired and checked by reading, not compiled, not run on hardware
//
#include <hip/hip_runtime.h>

typedef __attribute__((ext_vector_type(16))) _Float16 v16h;
typedef __attribute__((ext_vector_type(8)))  _Float16 v8h;
typedef __attribute__((ext_vector_type(4)))  _Float16 v4h;
typedef __attribute__((ext_vector_type(8)))  float    v8f;

#define HD    8
#define PS    16
#define NTOK  1024        // (512/16)^2
#define BATCH 2
#define HIMG  512
#define WIMG  512

__device__ __forceinline__ v16h cat8(v8h lo, v8h hi) {
  return __builtin_shufflevector(lo, hi, 0, 1, 2, 3, 4, 5, 6, 7,
                                         8, 9, 10, 11, 12, 13, 14, 15);
}

// Zero a vector unless `keep`, via bit-AND (no exec-mask branching).
__device__ __forceinline__ v8h maskz8(v8h v, bool keep) {
  unsigned m = keep ? 0xFFFFFFFFu : 0u;
  union { v8h h; unsigned u[4]; } x;
  x.h = v;
#pragma unroll
  for (int i = 0; i < 4; ++i) x.u[i] &= m;
  return x.h;
}
__device__ __forceinline__ v16h maskz16(v16h v, bool keep) {
  unsigned m = keep ? 0xFFFFFFFFu : 0u;
  union { v16h h; unsigned u[8]; } x;
  x.h = v;
#pragma unroll
  for (int i = 0; i < 8; ++i) x.u[i] &= m;
  return x.h;
}

// ---------------------------------------------------------------------------
// Patchify: x[B,1,512,512] -> p_f16[B*1024, 256]; 4 elements per thread.
// ---------------------------------------------------------------------------
__global__ void k_patchify(const float* __restrict__ x, _Float16* __restrict__ p) {
  int q = blockIdx.x * blockDim.x + threadIdx.x;       // over B*NTOK*64 quads
  if (q >= BATCH * NTOK * 64) return;
  int f4 = q & 63;
  int t  = (q >> 6) & (NTOK - 1);
  int b  = q >> 16;                                    // NTOK*64 = 2^16
  int i  = f4 >> 2;
  int j4 = (f4 & 3) * 4;
  const float4 v = *(const float4*)(x + (size_t)b * HIMG * WIMG +
                                    ((t >> 5) * PS + i) * WIMG + (t & 31) * PS + j4);
  v4h o = {(_Float16)v.x, (_Float16)v.y, (_Float16)v.z, (_Float16)v.w};
  *(v4h*)(p + (size_t)q * 4) = o;
}

// ---------------------------------------------------------------------------
// Pack W[Ktot,Nv] (f32) into WMMA B-layout tiles: packed[(c*32+lane)*16+e]
// with K = c*32 + (lane>>4)*16 + e, N = lane&15. Zero-padded.
// ---------------------------------------------------------------------------
__global__ void k_packW(const float* __restrict__ W, int Ktot, int Nv,
                        _Float16* __restrict__ packed, int nchunks) {
  int idx = blockIdx.x * blockDim.x + threadIdx.x;
  if (idx >= nchunks * 512) return;
  int e = idx & 15;
  int lane = (idx >> 4) & 31;
  int c = idx >> 9;
  int kk = c * 32 + (lane >> 4) * 16 + e;
  int n = lane & 15;
  float v = (kk < Ktot && n < Nv) ? W[kk * Nv + n] : 0.0f;
  packed[idx] = (_Float16)v;
}

// ---------------------------------------------------------------------------
// GEMM+bias via WMMA: out[M,NV] = A_f16[M,KTOT] @ Wpk(B-layout) + bias.
// One wave per 16-row tile.
// TRANS: write out[b][d][tok] (16-row padded); row NV is filled with 1.0
// (ones-column used by attention to accumulate softmax denominators).
// ---------------------------------------------------------------------------
template <int KTOT, int NV, bool TRANS>
__global__ void k_proj(const _Float16* __restrict__ A,
                       const _Float16* __restrict__ Wpk,
                       const float* __restrict__ bias,
                       _Float16* __restrict__ out, int Mtot) {
  int lane = threadIdx.x & 31;
  int m0 = blockIdx.x * 16;
  if (m0 >= Mtot) return;
  int r  = lane & 15;
  int hl = lane >> 4;
  int n  = lane & 15;
  v8h zz = {};
  v8f c = {};
  constexpr int NCH = (KTOT + 31) / 32;
#pragma unroll
  for (int ch = 0; ch < NCH; ++ch) {
    const _Float16* arow = A + (size_t)(m0 + r) * KTOT + ch * 32;
    v16h a;
    if constexpr (KTOT % 32 == 0) {
      v8h lo = *(const v8h*)(arow + hl * 16);
      v8h hi = *(const v8h*)(arow + hl * 16 + 8);
      a = cat8(lo, hi);
    } else {                         // KTOT==8: only hl==0, e<8 valid
      v8h lo = maskz8(*(const v8h*)(arow), hl == 0);
      a = cat8(lo, zz);
    }
    v16h bm = *(const v16h*)(Wpk + (size_t)(ch * 32 + lane) * 16);
    c = __builtin_amdgcn_wmma_f32_16x16x32_f16(false, a, false, bm, (short)0, c,
                                               false, false);
  }
  float bn = (n < NV) ? bias[n] : 0.0f;
  if constexpr (TRANS) {
    if (n <= NV) {
#pragma unroll
      for (int v = 0; v < 8; ++v) {
        int row = m0 + hl * 8 + v;
        _Float16 o = (n == NV) ? (_Float16)1.0f : (_Float16)(c[v] + bn);
        out[((size_t)(row >> 10) * 16 + n) * NTOK + (row & (NTOK - 1))] = o;
      }
    }
  } else {
    if (n < NV) {
#pragma unroll
      for (int v = 0; v < 8; ++v) {
        int row = m0 + hl * 8 + v;
        out[(size_t)row * NV + n] = (_Float16)(c[v] + bn);
      }
    }
  }
}

// ---------------------------------------------------------------------------
// Attention, N=1024 tokens, d=8, no-max softmax (scores are tiny: exact by
// shift invariance). V transposed+padded: Vt[b][16][tok], row 8 = ones so
// acc column 8 accumulates the softmax denominator via the same WMMA.
// One wave per 16-query tile; single-wave blocks -> LDS in-order, no barriers.
// ---------------------------------------------------------------------------
__global__ void k_attn_wmma(const _Float16* __restrict__ Qg,
                            const _Float16* __restrict__ Kg,
                            const _Float16* __restrict__ Vtg,
                            float* __restrict__ outF32, _Float16* __restrict__ outF16) {
  __shared__ __align__(32) _Float16 pT[16 * 32];   // P tile [query][key]
  int lane = threadIdx.x & 31;
  int b  = blockIdx.y;
  int m0 = blockIdx.x * 16;
  const _Float16* Q  = Qg  + (size_t)b * NTOK * HD;
  const _Float16* K  = Kg  + (size_t)b * NTOK * HD;
  const _Float16* Vt = Vtg + (size_t)b * 16 * NTOK;
  int r  = lane & 15;
  int hl = lane >> 4;
  int n  = lane & 15;
  v8h zz = {};

  // Q tile in A-layout, 1/sqrt(d) folded in (packed f16 multiply).
  v8h qv = maskz8(*(const v8h*)(Q + (size_t)(m0 + r) * HD), hl == 0);
  const _Float16 hscale = (_Float16)0.35355339059327373f;
#pragma unroll
  for (int i = 0; i < 8; ++i) qv[i] = qv[i] * hscale;
  v16h aq = cat8(qv, zz);

  v8f acc = {};                                    // cols 0..7 = O, col 8 = sum

  for (int kc = 0; kc < NTOK; kc += 32) {
    // B tiles for K^T: contraction = d (8 of 32); unconditional b128 loads,
    // upper half-wave zeroed by bit-mask (addresses valid for all lanes).
    v8h kv0 = maskz8(*(const v8h*)(K + (size_t)(kc + n) * HD), hl == 0);
    v8h kv1 = maskz8(*(const v8h*)(K + (size_t)(kc + 16 + n) * HD), hl == 0);
    v16h bk0 = cat8(kv0, zz);
    v16h bk1 = cat8(kv1, zz);
    v8f z8 = {};
    v8f s0 = __builtin_amdgcn_wmma_f32_16x16x32_f16(false, aq, false, bk0, (short)0,
                                                    z8, false, false);
    v8f s1 = __builtin_amdgcn_wmma_f32_16x16x32_f16(false, aq, false, bk1, (short)0,
                                                    z8, false, false);
    // p = exp(s): no reductions needed (denominator via ones-column of V).
#pragma unroll
    for (int v = 0; v < 8; ++v) {
      pT[(hl * 8 + v) * 32 + n]      = (_Float16)__expf(s0[v]);  // keys kc..+15
      pT[(hl * 8 + v) * 32 + 16 + n] = (_Float16)__expf(s1[v]);  // keys kc+16..+31
    }
    // Single-wave block: LDS ops are in-order (DScnt), no barrier needed.
    v8h plo = *(const v8h*)(pT + r * 32 + hl * 16);
    v8h phi = *(const v8h*)(pT + r * 32 + hl * 16 + 8);
    v16h ap = cat8(plo, phi);
    // V B-tile (dims 0..7 = V, dim 8 = ones, 9..15 masked to zero).
    v16h bv = maskz16(*(const v16h*)(Vt + (size_t)n * NTOK + kc + hl * 16),
                      n <= HD);
    acc = __builtin_amdgcn_wmma_f32_16x16x32_f16(false, ap, false, bv, (short)0, acc,
                                                 false, false);
  }

  // Column 8 of each half holds the row sums; broadcast and normalize.
#pragma unroll
  for (int v = 0; v < 8; ++v) {
    float l = __shfl(acc[v], hl * 16 + 8, 32);
    float o = acc[v] / l;
    if (n < HD) {
      int row = m0 + hl * 8 + v;
      outF32[((size_t)b * NTOK + row) * HD + n] = o;
      outF16[((size_t)b * NTOK + row) * HD + n] = (_Float16)o;
    }
  }
}

// ---------------------------------------------------------------------------
// LayerNorm + 8->3 head + bounded softplus; atomic max of (sx,sy) into ws.
// ---------------------------------------------------------------------------
__global__ void k_init_max(int* mb) {
  if (threadIdx.x == 0 && blockIdx.x == 0) *mb = 0;    // 0.0f bits
}

__global__ void k_head(const float* __restrict__ sp, const float* __restrict__ g,
                       const float* __restrict__ bta, const float* __restrict__ Wp,
                       const float* __restrict__ bp, float* __restrict__ sig,
                       int* __restrict__ maxbits) {
  int t = blockIdx.x * blockDim.x + threadIdx.x;
  if (t >= BATCH * NTOK) return;
  float a[HD];
  float mu = 0.0f;
#pragma unroll
  for (int i = 0; i < HD; ++i) { a[i] = sp[(size_t)t * HD + i]; mu += a[i]; }
  mu *= (1.0f / HD);
  float var = 0.0f;
#pragma unroll
  for (int i = 0; i < HD; ++i) { float d = a[i] - mu; var += d * d; }
  var *= (1.0f / HD);
  float rstd = rsqrtf(var + 1e-5f);
  float xn[HD];
#pragma unroll
  for (int i = 0; i < HD; ++i) xn[i] = (a[i] - mu) * rstd * g[i] + bta[i];
  float smax = 0.0f;
#pragma unroll
  for (int j = 0; j < 3; ++j) {
    float y = bp[j];
#pragma unroll
    for (int i = 0; i < HD; ++i) y += xn[i] * Wp[i * 3 + j];
    float s = (y > 15.0f) ? y : log1pf(__expf(y));     // softplus
    s = fminf(s, 6.0f) + 1e-6f;                        // bounded + eps
    sig[(size_t)t * 3 + j] = s;
    if (j < 2) smax = fmaxf(smax, s);
  }
  atomicMax(maxbits, __float_as_int(smax));            // all values > 0
}

// ---------------------------------------------------------------------------
// Bilateral filter with data-dependent kernel size (read from workspace).
// ---------------------------------------------------------------------------
__device__ __forceinline__ int reflect_idx(int i, int nmax) {
  if (i < 0) i = -i;
  if (i >= nmax) i = 2 * nmax - 2 - i;
  return i;
}

template <int KS>
__device__ __forceinline__ float bilat_fixed(const float* __restrict__ Xb,
                                             int xq, int yq, float i2x, float i2y,
                                             float i2r, float center) {
  constexpr int P = KS / 2;
  float num = 0.0f, den = 0.0f;
#pragma unroll
  for (int i = 0; i < KS; ++i) {
    int dy = i - P;
    int ry = reflect_idx(yq + dy, HIMG);
    float wy = (float)(dy * dy) * i2y;
#pragma unroll
    for (int j = 0; j < KS; ++j) {
      int dx = j - P;
      int rx = reflect_idx(xq + dx, WIMG);
      float v = Xb[ry * WIMG + rx];
      float d = center - v;
      float w = __expf(-((float)(dx * dx) * i2x + wy + d * d * i2r));
      num += w * v;
      den += w;
    }
  }
  return num / (den + 1e-8f);
}

__device__ float bilat_dyn(const float* __restrict__ Xb, int xq, int yq,
                           float i2x, float i2y, float i2r, float center, int ksz) {
  int p = ksz >> 1;
  float num = 0.0f, den = 0.0f;
  for (int i = 0; i < ksz; ++i) {
    int dy = i - p;
    int ry = reflect_idx(yq + dy, HIMG);
    float wy = (float)(dy * dy) * i2y;
    for (int j = 0; j < ksz; ++j) {
      int dx = j - p;
      int rx = reflect_idx(xq + dx, WIMG);
      float v = Xb[ry * WIMG + rx];
      float d = center - v;
      float w = __expf(-((float)(dx * dx) * i2x + wy + d * d * i2r));
      num += w * v;
      den += w;
    }
  }
  return num / (den + 1e-8f);
}

__global__ void k_filter(const float* __restrict__ X, const float* __restrict__ sig,
                         const int* __restrict__ maxbits, float* __restrict__ out) {
  int xq = blockIdx.x * blockDim.x + threadIdx.x;
  int yq = blockIdx.y * blockDim.y + threadIdx.y;
  int b  = blockIdx.z;
  if (xq >= WIMG || yq >= HIMG) return;
  float m = __int_as_float(*maxbits);
  int ksz = 2 * (int)ceilf(m) + 1;
  if ((ksz & 1) == 0) ksz++;
  if (ksz < 1) ksz = 1;
  if (ksz > 41) ksz = 41;
  const float* Xb = X + (size_t)b * HIMG * WIMG;
  int pi = (yq >> 4) * 32 + (xq >> 4);                 // nearest-upsampled sigma
  const float* sg = sig + ((size_t)b * NTOK + pi) * 3;
  float sx = sg[0], sy = sg[1], sr = sg[2];
  float i2x = 0.5f / (sx * sx);                        // x = column offset
  float i2y = 0.5f / (sy * sy);                        // y = row offset
  float i2r = 0.5f / (sr * sr);
  float center = Xb[yq * WIMG + xq];
  float r;
  if (ksz == 11) {                                     // expected case (sig~4.5)
    r = bilat_fixed<11>(Xb, xq, yq, i2x, i2y, i2r, center);
  } else {
    r = bilat_dyn(Xb, xq, yq, i2x, i2y, i2r, center, ksz);
  }
  out[(size_t)b * HIMG * WIMG + yq * WIMG + xq] = r;
}

// ---------------------------------------------------------------------------
extern "C" void kernel_launch(void* const* d_in, const int* in_sizes, int n_in,
                              void* d_out, int out_size, void* d_ws, size_t ws_size,
                              hipStream_t stream) {
  const float* x    = (const float*)d_in[0];
  const float* Wq   = (const float*)d_in[1];  const float* bq  = (const float*)d_in[2];
  const float* Wk   = (const float*)d_in[3];  const float* bk  = (const float*)d_in[4];
  const float* Wv   = (const float*)d_in[5];  const float* bv  = (const float*)d_in[6];
  const float* Wsq  = (const float*)d_in[7];  const float* bsq = (const float*)d_in[8];
  const float* Wsk  = (const float*)d_in[9];  const float* bsk = (const float*)d_in[10];
  const float* Wsv  = (const float*)d_in[11]; const float* bsv = (const float*)d_in[12];
  const float* ln_g = (const float*)d_in[13]; const float* ln_b = (const float*)d_in[14];
  const float* Wp   = (const float*)d_in[15]; const float* bp   = (const float*)d_in[16];

  char* ws = (char*)d_ws;
  size_t off = 0;
  auto walloc = [&](size_t bytes) -> void* {
    void* p = ws + off;
    off = (off + bytes + 255) & ~(size_t)255;
    return p;
  };
  const int M = BATCH * NTOK;                          // 2048 tokens
  _Float16* pF   = (_Float16*)walloc((size_t)M * 256 * sizeof(_Float16));
  _Float16* Qf   = (_Float16*)walloc((size_t)M * HD * sizeof(_Float16));
  _Float16* Kf   = (_Float16*)walloc((size_t)M * HD * sizeof(_Float16));
  _Float16* Vt   = (_Float16*)walloc((size_t)BATCH * 16 * NTOK * sizeof(_Float16));
  float*    f32a = (float*)   walloc((size_t)M * HD * sizeof(float));
  _Float16* ff16 = (_Float16*)walloc((size_t)M * HD * sizeof(_Float16));
  _Float16* Qs   = (_Float16*)walloc((size_t)M * HD * sizeof(_Float16));
  _Float16* Ks   = (_Float16*)walloc((size_t)M * HD * sizeof(_Float16));
  _Float16* Vst  = (_Float16*)walloc((size_t)BATCH * 16 * NTOK * sizeof(_Float16));
  float*    sp32 = (float*)   walloc((size_t)M * HD * sizeof(float));
  _Float16* sp16 = (_Float16*)walloc((size_t)M * HD * sizeof(_Float16));
  float*    sig  = (float*)   walloc((size_t)M * 3 * sizeof(float));
  int*   maxbits = (int*)     walloc(sizeof(int));
  _Float16* WqP  = (_Float16*)walloc(8 * 512 * sizeof(_Float16));  // K=256 packs
  _Float16* WkP  = (_Float16*)walloc(8 * 512 * sizeof(_Float16));
  _Float16* WvP  = (_Float16*)walloc(8 * 512 * sizeof(_Float16));
  _Float16* WsqP = (_Float16*)walloc(1 * 512 * sizeof(_Float16));  // K=8 packs
  _Float16* WskP = (_Float16*)walloc(1 * 512 * sizeof(_Float16));
  _Float16* WsvP = (_Float16*)walloc(1 * 512 * sizeof(_Float16));

  // 0) pack weights into WMMA B-layout
  k_packW<<<16, 256, 0, stream>>>(Wq, 256, HD, WqP, 8);
  k_packW<<<16, 256, 0, stream>>>(Wk, 256, HD, WkP, 8);
  k_packW<<<16, 256, 0, stream>>>(Wv, 256, HD, WvP, 8);
  k_packW<<<2, 256, 0, stream>>>(Wsq, HD, HD, WsqP, 1);
  k_packW<<<2, 256, 0, stream>>>(Wsk, HD, HD, WskP, 1);
  k_packW<<<2, 256, 0, stream>>>(Wsv, HD, HD, WsvP, 1);

  // 1) patchify + f16 cast (float4 -> 4x f16 per thread)
  k_patchify<<<(BATCH * NTOK * 64) / 256, 256, 0, stream>>>(x, pF);

  // 2) QKV projections (256 -> 8) via WMMA; V transposed + ones row
  k_proj<256, HD, false><<<M / 16, 32, 0, stream>>>(pF, WqP, bq, Qf, M);
  k_proj<256, HD, false><<<M / 16, 32, 0, stream>>>(pF, WkP, bk, Kf, M);
  k_proj<256, HD, true ><<<M / 16, 32, 0, stream>>>(pF, WvP, bv, Vt, M);

  // 3) attention #1
  k_attn_wmma<<<dim3(NTOK / 16, BATCH), 32, 0, stream>>>(Qf, Kf, Vt, f32a, ff16);

  // 4) second-stage projections (8 -> 8) via WMMA; V transposed + ones row
  k_proj<HD, HD, false><<<M / 16, 32, 0, stream>>>(ff16, WsqP, bsq, Qs, M);
  k_proj<HD, HD, false><<<M / 16, 32, 0, stream>>>(ff16, WskP, bsk, Ks, M);
  k_proj<HD, HD, true ><<<M / 16, 32, 0, stream>>>(ff16, WsvP, bsv, Vst, M);

  // 5) attention #2
  k_attn_wmma<<<dim3(NTOK / 16, BATCH), 32, 0, stream>>>(Qs, Ks, Vst, sp32, sp16);

  // 6) LN + head + bounded softplus + global max for kernel size
  k_init_max<<<1, 1, 0, stream>>>(maxbits);
  k_head<<<(M + 255) / 256, 256, 0, stream>>>(sp32, ln_g, ln_b, Wp, bp, sig, maxbits);

  // 7) bilateral filter with dynamic kernel size (k==11 path fully unrolled)
  k_filter<<<dim3(WIMG / 16, HIMG / 16, BATCH), dim3(16, 16), 0, stream>>>(
      x, sig, maxbits, (float*)d_out);
}
